// FactorModel_47313359733391
// MI455X (gfx1250) — matrix operand order repaired and checked by reading, not compiled, not run on hardware
//
#include <hip/hip_runtime.h>
#include <hip/hip_bf16.h>

typedef __attribute__((ext_vector_type(16))) _Float16 v16h;
typedef __attribute__((ext_vector_type(8)))  float    v8f;

#define WMMA_F16(a, b, c) \
  __builtin_amdgcn_wmma_f32_16x16x32_f16(false, (a), false, (b), (short)0, (c), false, false)

// sched_group_barrier masks
#define SG_MFMA 0x008   // MFMA/WMMA class
#define SG_DSRD 0x100   // DS read

// ---------------- problem sizes ----------------
constexpr int BATCH  = 256;
constexpr int TPREV  = 128;
constexpr int TCUR   = 129;
constexpr int NCOV   = 30;
constexpr int NTREAT = 57;
constexpr int NCONF  = 5;
constexpr int DIN    = 87;           // external lstm input
constexpr int KTOT   = 224;          // 92 (x|z) padded to 96, + 128 (h)
constexpr int KF     = KTOT / 32;    // 7 k-fragments per gate GEMM
constexpr long NROWS = (long)TCUR * BATCH;          // 33024
constexpr long PRED_ELEMS = NROWS * NTREAT;         // 1882368

// ---------------- fragment loaders ----------------
__device__ __forceinline__ v16h ld_frag2(const _Float16* p0, const _Float16* p1) {
  union { uint4 q[2]; v16h v; } u;
  u.q[0] = *(const uint4*)p0;
  u.q[1] = *(const uint4*)p1;
  return u.v;
}
__device__ __forceinline__ v16h ld_frag32(const _Float16* p) {
  union { uint4 q[2]; v16h v; } u;
  u.q[0] = ((const uint4*)p)[0];
  u.q[1] = ((const uint4*)p)[1];
  return u.v;
}

__device__ __forceinline__ float fast_sig(float x) {
  x = fminf(30.f, fmaxf(-30.f, x));
  return 1.f / (1.f + __expf(-x));
}
__device__ __forceinline__ float fast_tanh(float x) {
  x = fminf(20.f, fmaxf(-20.f, x));
  float e = __expf(-2.f * x);
  return (1.f - e) / (1.f + e);
}

// =====================================================================
// Kernel 1: independent-batch LSTM scan. 16 blocks x 16 batch rows.
// 8 waves/block; wave w owns hidden columns [16w, 16w+16) and computes
// its own i/f/g/o 16x16 tiles. Weights live in LDS as pre-swizzled f16
// WMMA B-fragments. The gate GEMM is a distance-2 software pipeline
// enforced with sched_group_barrier so WMMAs overlap DS-read latency.
// =====================================================================
__global__ __launch_bounds__(256) void lstm_scan_kernel(
    const float* __restrict__ pc,  const float* __restrict__ pt,
    const float* __restrict__ init_input,
    const float* __restrict__ h0,  const float* __restrict__ c0,
    const float* __restrict__ z0,
    const float* __restrict__ W_ih, const float* __restrict__ W_hh,
    const float* __restrict__ b_ih, const float* __restrict__ b_hh,
    const float* __restrict__ W_z,  const float* __restrict__ b_z,
    float* __restrict__ zs_out)
{
  extern __shared__ __align__(16) char smem[];
  _Float16* A    = (_Float16*)smem;            // [16][224] row-major activations
  _Float16* Bg   = A + 16 * KTOT;              // 224 frags * 512 halves (gate weights)
  _Float16* Bz   = Bg + 224 * 512;             // 4 frags (W_z, cols 5..15 zero)
  float*    bias = (float*)(Bz + 4 * 512);     // 512
  float*    bzs  = bias + 512;                 // 8

  const int tid   = threadIdx.x;
  const int wv    = tid >> 5;
  const int ln    = tid & 31;
  const int col   = ln & 15;
  const int half  = ln >> 4;
  const int bbase = blockIdx.x * 16;

  // ---- one-time weight staging (f32 global -> f16 LDS, B-frag order) ----
  for (int i = tid; i < 512; i += 256) bias[i] = b_ih[i] + b_hh[i];
  if (tid < 8) bzs[tid] = (tid < NCONF) ? b_z[tid] : 0.f;

  for (int idx = tid; idx < 224 * 32; idx += 256) {   // (frag, lane) pairs
    int f = idx >> 5, l = idx & 31;
    int kf = f >> 5, nt = f & 31;
    int kbase = kf * 32 + ((l >> 4) << 4);
    int n = nt * 16 + (l & 15);
    _Float16* dst = Bg + idx * 16;
    #pragma unroll
    for (int j = 0; j < 16; ++j) {
      int k = kbase + j;
      float v;
      if (k < 92)      v = W_ih[n * 92 + k];
      else if (k < 96) v = 0.f;                       // pad
      else             v = W_hh[n * 128 + (k - 96)];
      dst[j] = (_Float16)v;
    }
  }
  for (int idx = tid; idx < 4 * 32; idx += 256) {     // W_z frags
    int zf = idx >> 5, l = idx & 31;
    int kbase = zf * 32 + ((l >> 4) << 4);
    int n = l & 15;
    _Float16* dst = Bz + idx * 16;
    #pragma unroll
    for (int j = 0; j < 16; ++j)
      dst[j] = (n < NCONF) ? (_Float16)W_z[n * 128 + kbase + j] : (_Float16)0.f;
  }
  // initial A: zeros for x, z0 at cols 87..91, pad, h0 at cols 96..223
  for (int i = tid; i < 16 * KTOT; i += 256) {
    int m = i / KTOT, k = i % KTOT;
    float v;
    if (k < DIN)      v = 0.f;
    else if (k < 92)  v = z0[k - DIN];
    else if (k < 96)  v = 0.f;
    else              v = h0[k - 96];
    A[m * KTOT + k] = (_Float16)v;
  }
  float creg[8];
  {
    float cv = c0[wv * 16 + col];
    #pragma unroll
    for (int r = 0; r < 8; ++r) creg[r] = cv;
  }
  __syncthreads();

  const float bi = bias[  0 + wv * 16 + col];
  const float bf = bias[128 + wv * 16 + col];
  const float bg = bias[256 + wv * 16 + col];
  const float bo = bias[384 + wv * 16 + col];
  const _Float16* arow = A + (ln & 15) * KTOT;        // lane row for A frags
  const int mb = half * 8;                            // accumulator row base
  // x staging: thread (tid>>4) owns row m, strided columns (no div/mod in loop)
  const int xm  = tid >> 4;
  const int xj0 = tid & 15;
  _Float16* xdst = A + xm * KTOT;

  for (int t = 0; t < TCUR; ++t) {
    // ---- stage x_t into A cols [0,87) ----
    if (t == 0) {
      for (int j = xj0; j < DIN; j += 16) xdst[j] = (_Float16)init_input[j];
    } else {
      const int boff = ((bbase + xm) * TPREV + (t - 1));
      const float* pcb = pc + boff * NCOV;
      const float* ptb = pt + boff * NTREAT;
      for (int j = xj0; j < DIN; j += 16)
        xdst[j] = (_Float16)((j < NCOV) ? pcb[j] : ptb[j - NCOV]);
    }
    __syncthreads();                                  // B1: x,z visible

    // ---- gates = A(16x224) @ Wc^T: distance-2 pipelined, 28 WMMAs ----
    // Preamble group: 14 A-frag DS reads + stage0 + stage1 B reads (30).
    v16h afrag[KF];
    #pragma unroll
    for (int kf = 0; kf < KF; ++kf) {
      const _Float16* p0 = arow + kf * 32 + half * 16;
      afrag[kf] = ld_frag2(p0, p0 + 8);
    }
    v16h bbuf[2][4];
    #pragma unroll
    for (int s = 0; s < 2; ++s)
      #pragma unroll
      for (int g = 0; g < 4; ++g)
        bbuf[s][g] = ld_frag32(Bg + ((s * 32 + g * 8 + wv) * 32 + ln) * 16);
    __builtin_amdgcn_sched_group_barrier(SG_DSRD, 30, 0);

    v8f acc[4];
    #pragma unroll
    for (int g = 0; g < 4; ++g) acc[g] = v8f{};
    #pragma unroll
    for (int kf = 0; kf < KF; ++kf) {
      // consume stage kf ...
      #pragma unroll
      for (int g = 0; g < 4; ++g)
        acc[g] = WMMA_F16(afrag[kf], bbuf[kf & 1][g], acc[g]);
      // ... then prefetch stage kf+2 (clamped; keeps 8-load groups regular)
      const int kfp = (kf + 2 < KF) ? (kf + 2) : (KF - 1);
      #pragma unroll
      for (int g = 0; g < 4; ++g)
        bbuf[kf & 1][g] = ld_frag32(Bg + ((kfp * 32 + g * 8 + wv) * 32 + ln) * 16);
      __builtin_amdgcn_sched_group_barrier(SG_MFMA, 4, 0);
      __builtin_amdgcn_sched_group_barrier(SG_DSRD, 8, 0);
    }
    // ---- cell update (f32, in registers) ----
    float hval[8];
    #pragma unroll
    for (int r = 0; r < 8; ++r) {
      float iv = acc[0][r] + bi;
      float fv = acc[1][r] + bf;
      float gv = acc[2][r] + bg;
      float ov = acc[3][r] + bo;
      float cn = fast_sig(fv) * creg[r] + fast_sig(iv) * fast_tanh(gv);
      creg[r]  = cn;
      hval[r]  = fast_sig(ov) * fast_tanh(cn);
    }
    __syncthreads();                                  // B2: A reads done
    {
      int hcol = 96 + wv * 16 + col;
      #pragma unroll
      for (int r = 0; r < 8; ++r)
        A[(mb + r) * KTOT + hcol] = (_Float16)hval[r];
    }
    __syncthreads();                                  // B3: h visible

    // ---- z = h @ W_z^T (wave 0 only; uniform branch keeps EXEC full) ----
    if (wv == 0) {
      v8f za = v8f{};
      #pragma unroll
      for (int zf = 0; zf < 4; ++zf) {
        const _Float16* p0 = arow + 96 + zf * 32 + half * 16;
        v16h ah = ld_frag2(p0, p0 + 8);
        v16h bfr = ld_frag32(Bz + (zf * 32 + ln) * 16);
        za = WMMA_F16(ah, bfr, za);
      }
      if (col < NCONF) {
        float bzv = bzs[col];
        #pragma unroll
        for (int r = 0; r < 8; ++r) {
          float zv = za[r] + bzv;
          int m = mb + r;
          A[m * KTOT + DIN + col] = (_Float16)zv;
          zs_out[((long)t * BATCH + bbase + m) * NCONF + col] = zv;
        }
      }
    }
    // next iteration's B1 makes the z write visible to all waves
  }
}

// =====================================================================
// Kernel 2: per-treatment decoder MLP. grid = (33024/128, 57).
// Each block: one treatment, 8 waves x one 16-row tile.
// h1 = leaky_relu(mt(16x35pad64) @ W1[t]); pred = h1 . W2[t] + b2.
// =====================================================================
__global__ __launch_bounds__(256) void decoder_kernel(
    const float* __restrict__ zs,  const float* __restrict__ cc,
    const float* __restrict__ dW1, const float* __restrict__ db1,
    const float* __restrict__ dW2, const float* __restrict__ db2,
    float* __restrict__ preds)
{
  __shared__ __align__(16) _Float16 Bw[16 * 512];     // 2 kfrags x 8 ntiles
  __shared__ __align__(16) _Float16 Ast[8 * 16 * 64]; // per-wave A staging
  __shared__ float w2s[128], b1s[128];

  const int tid  = threadIdx.x;
  const int wv   = tid >> 5;
  const int ln   = tid & 31;
  const int col  = ln & 15;
  const int half = ln >> 4;
  const int treat = blockIdx.y;
  const long rowg = ((long)blockIdx.x * 8 + wv) * 16;

  for (int i = tid; i < 128; i += 256) {
    w2s[i] = dW2[treat * 128 + i];
    b1s[i] = db1[treat * 128 + i];
  }
  for (int idx = tid; idx < 16 * 32; idx += 256) {    // W1 -> B frags
    int f = idx >> 5, l = idx & 31;
    int kf = f >> 3, nt = f & 7;
    int kbase = kf * 32 + ((l >> 4) << 4);
    int ff = nt * 16 + (l & 15);
    _Float16* dst = Bw + idx * 16;
    #pragma unroll
    for (int j = 0; j < 16; ++j) {
      int k = kbase + j;
      dst[j] = (k < NCONF + NCOV)
                 ? (_Float16)dW1[((long)treat * 35 + k) * 128 + ff]
                 : (_Float16)0.f;
    }
  }
  // A staging: [z(5) | cov(30) | zero pad to 64] per row, f16
  for (int i = ln; i < 16 * 64; i += 32) {
    int m = i >> 6, j = i & 63;
    long n = rowg + m;
    float v;
    if (j < NCONF) v = zs[n * NCONF + j];
    else if (j < 35) {
      long ts = n >> 8, b = n & 255;                  // n = ts*256 + b
      v = cc[(b * TCUR + ts) * NCOV + (j - NCONF)];
    } else v = 0.f;
    Ast[(wv * 16 + m) * 64 + j] = (_Float16)v;
  }
  __syncthreads();

  const _Float16* arow = Ast + (wv * 16 + (ln & 15)) * 64;
  v16h af0 = ld_frag2(arow +      half * 16, arow +  8 + half * 16);
  v16h af1 = ld_frag2(arow + 32 + half * 16, arow + 40 + half * 16);

  float rsum[8];
  #pragma unroll
  for (int r = 0; r < 8; ++r) rsum[r] = 0.f;

  // distance-2 pipelined B fragments across the 8 output tiles
  v16h bbuf[2][2];
  #pragma unroll
  for (int s = 0; s < 2; ++s) {
    bbuf[s][0] = ld_frag32(Bw + ((0 * 8 + s) * 32 + ln) * 16);
    bbuf[s][1] = ld_frag32(Bw + ((1 * 8 + s) * 32 + ln) * 16);
  }
  __builtin_amdgcn_sched_group_barrier(SG_DSRD, 8, 0); // 4 A + 4 B preamble

  #pragma unroll
  for (int nt = 0; nt < 8; ++nt) {
    v8f a = v8f{};
    a = WMMA_F16(af0, bbuf[nt & 1][0], a);
    a = WMMA_F16(af1, bbuf[nt & 1][1], a);
    const int ntp = (nt + 2 < 8) ? (nt + 2) : 7;
    bbuf[nt & 1][0] = ld_frag32(Bw + ((0 * 8 + ntp) * 32 + ln) * 16);
    bbuf[nt & 1][1] = ld_frag32(Bw + ((1 * 8 + ntp) * 32 + ln) * 16);
    __builtin_amdgcn_sched_group_barrier(SG_MFMA, 2, 0);
    __builtin_amdgcn_sched_group_barrier(SG_DSRD, 2, 0);
    float b1v = b1s[nt * 16 + col];
    float w2v = w2s[nt * 16 + col];
    #pragma unroll
    for (int r = 0; r < 8; ++r) {
      float hv = a[r] + b1v;
      hv = hv > 0.f ? hv : 0.01f * hv;                // LeakyReLU(0.01)
      rsum[r] += hv * w2v;
    }
  }
  // reduce over the 16 feature-lanes in each half-wave
  #pragma unroll
  for (int r = 0; r < 8; ++r) {
    float s = rsum[r];
    s += __shfl_xor(s, 1, 16);
    s += __shfl_xor(s, 2, 16);
    s += __shfl_xor(s, 4, 16);
    s += __shfl_xor(s, 8, 16);
    rsum[r] = s;
  }
  if (col == 0) {
    float b2v = db2[treat];
    int mbase = half * 8;
    #pragma unroll
    for (int r = 0; r < 8; ++r) {
      long n = rowg + mbase + r;
      preds[n * NTREAT + treat] = rsum[r] + b2v;
    }
  }
}

// =====================================================================
extern "C" void kernel_launch(void* const* d_in, const int* in_sizes, int n_in,
                              void* d_out, int out_size, void* d_ws, size_t ws_size,
                              hipStream_t stream) {
  (void)in_sizes; (void)n_in; (void)out_size; (void)d_ws; (void)ws_size;
  const float* pc  = (const float*)d_in[0];
  const float* pt  = (const float*)d_in[1];
  const float* cc  = (const float*)d_in[2];
  const float* ii  = (const float*)d_in[3];
  const float* h0  = (const float*)d_in[4];
  const float* c0  = (const float*)d_in[5];
  const float* z0  = (const float*)d_in[6];
  const float* Wih = (const float*)d_in[7];
  const float* Whh = (const float*)d_in[8];
  const float* bih = (const float*)d_in[9];
  const float* bhh = (const float*)d_in[10];
  const float* Wz  = (const float*)d_in[11];
  const float* bz  = (const float*)d_in[12];
  const float* dW1 = (const float*)d_in[13];
  const float* db1 = (const float*)d_in[14];
  const float* dW2 = (const float*)d_in[15];
  const float* db2 = (const float*)d_in[16];

  float* out   = (float*)d_out;
  float* preds = out;                  // (33024, 57)
  float* zs    = out + PRED_ELEMS;     // confounders flat == zs flat (33024*5)

  size_t lds1 = (size_t)(16 * KTOT + 224 * 512 + 4 * 512) * sizeof(_Float16)
              + (512 + 8) * sizeof(float);            // ~243 KB (<= 320 KB/WG on CDNA5)
  (void)hipFuncSetAttribute((const void*)lstm_scan_kernel,
                            hipFuncAttributeMaxDynamicSharedMemorySize, (int)lds1);

  lstm_scan_kernel<<<16, 256, lds1, stream>>>(pc, pt, ii, h0, c0, z0,
                                              Wih, Whh, bih, bhh, Wz, bz, zs);

  dim3 grid2((unsigned)(NROWS / 128), NTREAT);        // (258, 57)
  decoder_kernel<<<grid2, 256, 0, stream>>>(zs, cc, dW1, db1, dW2, db2, preds);
}
